// HANLayer_52458730553688
// MI455X (gfx1250) — matrix operand order repaired and checked by reading, not compiled, not run on hardware
//
#include <hip/hip_runtime.h>
#include <hip/hip_bf16.h>
#include <math.h>

// Problem constants (match reference setup_inputs)
constexpr int   N_ = 50000;
constexpr int   E_ = 1600000;
constexpr int   M_ = 3;
constexpr int   D_ = 128;   // feature dim == hidden dim H

typedef __attribute__((ext_vector_type(16))) __bf16 v16bf;
typedef __attribute__((ext_vector_type(8)))  __bf16 v8bf;
typedef __attribute__((ext_vector_type(4)))  __bf16 v4bf;
typedef __attribute__((ext_vector_type(8)))  float  v8f;

// ---------------------------------------------------------------- zero ws
__global__ void k_zero(float* __restrict__ p, size_t n) {
    size_t i = (size_t)blockIdx.x * blockDim.x + threadIdx.x;
    size_t stride = (size_t)gridDim.x * blockDim.x;
    for (; i < n; i += stride) p[i] = 0.0f;
}

// ------------------------------------------------------------- degrees
__global__ void k_deg(const int* __restrict__ edges,
                      float* __restrict__ deg_src,
                      float* __restrict__ deg_dst) {
    long long tid = (long long)blockIdx.x * blockDim.x + threadIdx.x;
    if (tid >= (long long)M_ * E_) return;
    int m = (int)(tid / E_);
    int e = (int)(tid - (long long)m * E_);
    int s = edges[((size_t)m * 2 + 0) * E_ + e];
    int d = edges[((size_t)m * 2 + 1) * E_ + e];
    atomicAdd(&deg_src[(size_t)m * N_ + s], 1.0f);
    atomicAdd(&deg_dst[(size_t)m * N_ + d], 1.0f);
}

// deg -> rsqrt(max(deg,1)); deg_src and deg_dst are contiguous: one pass
__global__ void k_norm(float* __restrict__ deg, int n) {
    int i = blockIdx.x * blockDim.x + threadIdx.x;
    if (i >= n) return;
    deg[i] = rsqrtf(fmaxf(deg[i], 1.0f));
}

// --------------------------------------------- per-edge wave32 scatter
// one wave per (metapath, edge): 32 lanes x float4 = 128-float row
__global__ void k_scatter(const float* __restrict__ h,
                          const int*   __restrict__ edges,
                          const float* __restrict__ nsrc,  // [M][N] rsqrt out-deg
                          float*       __restrict__ z) {   // [N][M][D]
    const int lane = threadIdx.x & 31;
    const long long gw = (long long)blockIdx.x * (blockDim.x >> 5) + (threadIdx.x >> 5);
    if (gw >= (long long)M_ * E_) return;
    const int m = (int)(gw / E_);
    const int e = (int)(gw - (long long)m * E_);

    const int* esrc = edges + ((size_t)m * 2 + 0) * E_;
    const int* edst = edges + ((size_t)m * 2 + 1) * E_;
    // lookahead prefetch on the (streamed) edge lists -> global_prefetch_b8
    __builtin_prefetch(esrc + e + 4096, 0, 1);
    __builtin_prefetch(edst + e + 4096, 0, 1);

    const int src = esrc[e];
    const int dst = edst[e];
    const float ns = nsrc[(size_t)m * N_ + src];

    const float4 hv = *(const float4*)(h + (size_t)src * D_ + lane * 4);
    float* zp = z + ((size_t)dst * M_ + m) * D_ + lane * 4;
    atomicAdd(zp + 0, hv.x * ns);
    atomicAdd(zp + 1, hv.y * ns);
    atomicAdd(zp + 2, hv.z * ns);
    atomicAdd(zp + 3, hv.w * ns);
}

// ------------------------- scale rows by dst norm + emit bf16 copy of z
__global__ void k_scale_dst(float* __restrict__ z,
                            const float* __restrict__ ndst,  // [M][N]
                            __bf16* __restrict__ zbf) {      // [N*M][D] bf16
    size_t tid = (size_t)blockIdx.x * blockDim.x + threadIdx.x;
    const size_t total = (size_t)N_ * M_ * (D_ / 4);
    if (tid >= total) return;
    size_t row = tid / (D_ / 4);
    int m = (int)(row % M_);
    int n = (int)(row / M_);
    float s = ndst[(size_t)m * N_ + n];
    float4* zp = (float4*)z + tid;
    float4 v = *zp;
    v.x *= s; v.y *= s; v.z *= s; v.w *= s;
    *zp = v;
    v4bf o;
    o[0] = (__bf16)v.x; o[1] = (__bf16)v.y;
    o[2] = (__bf16)v.z; o[3] = (__bf16)v.w;
    *(v4bf*)(zbf + tid * 4) = o;
}

// ------------------------------------- WMMA attention-score GEMM + reduce
// W1^T staged once per block into LDS as bf16 (column-major, 32KB).
// Each wave: 16 rows of Zbf x W1 via v_wmma_f32_16x16x32_bf16; A fragments are
// two contiguous b128 loads, B fragments are one 32-byte LDS read.
// NOTE: the asm memory clobber in the tile loop stops LICM from hoisting all
// 32 loop-invariant B-fragment LDS loads (256 VGPRs) out of the loop, which
// previously caused scratch spills of every WMMA B operand.
__global__ void k_attn_score(const __bf16* __restrict__ zbf,
                             const float* __restrict__ W1,
                             const float* __restrict__ b1,
                             const float* __restrict__ W2,
                             float* __restrict__ score_sum) {  // [M]
    __shared__ __bf16 ldsW1T[D_ * D_];   // [col][row] bf16, 32 KB

    // Cooperative staging: coalesced float4 row reads, transposed bf16 stores.
    for (int i = threadIdx.x; i < (D_ * D_) / 4; i += blockDim.x) {
        int r  = i >> 5;            // row of W1 (K index), 32 float4 per row
        int c4 = (i & 31) * 4;      // starting column (H index)
        float4 w = *(const float4*)(W1 + (size_t)r * D_ + c4);
        ldsW1T[(c4 + 0) * D_ + r] = (__bf16)w.x;
        ldsW1T[(c4 + 1) * D_ + r] = (__bf16)w.y;
        ldsW1T[(c4 + 2) * D_ + r] = (__bf16)w.z;
        ldsW1T[(c4 + 3) * D_ + r] = (__bf16)w.w;
    }
    __syncthreads();

    const int lane = threadIdx.x & 31;
    const int wave = threadIdx.x >> 5;
    const int wavesPerBlock = blockDim.x >> 5;
    const int half = lane >> 4;          // K-half selector (A/B layouts)
    const int l16  = lane & 15;          // row (A) / col (B,C)
    const int nTiles = (N_ * M_) / 16;   // 9375

    for (int tile = blockIdx.x * wavesPerBlock + wave; tile < nTiles;
         tile += gridDim.x * wavesPerBlock) {
        // Prevent hoisting of loop-invariant LDS loads (defeats spill-causing LICM)
        asm volatile("" ::: "memory");

        const int rowBase = tile * 16;
        v8f acc[8] = {};                 // 8 N-tiles of 16 cols -> full 128

        const __bf16* zr = zbf + (size_t)(rowBase + l16) * D_;

        #pragma unroll
        for (int kt = 0; kt < 4; ++kt) { // K = 128 in chunks of 32
            const int kb = kt * 32;
            // A fragment: lane holds K run [kb+16h, +16)
            v8bf lo = *(const v8bf*)(zr + kb + 16 * half);
            v8bf hi = *(const v8bf*)(zr + kb + 16 * half + 8);
            v16bf a;
            #pragma unroll
            for (int i = 0; i < 8; ++i) { a[i] = lo[i]; a[8 + i] = hi[i]; }

            #pragma unroll
            for (int nt = 0; nt < 8; ++nt) {
                // B fragment: 16 contiguous bf16 down column nt*16+l16
                v16bf b = *(const v16bf*)(ldsW1T +
                            (size_t)(nt * 16 + l16) * D_ + kb + 16 * half);
                acc[nt] = __builtin_amdgcn_wmma_f32_16x16x32_bf16(
                    false, a, false, b, (short)0, acc[nt], false, false);
            }
        }

        // C layout: acc[nt][j], lane -> row = j + 8*half, col = nt*16 + l16
        float partial[8];
        #pragma unroll
        for (int j = 0; j < 8; ++j) partial[j] = 0.0f;
        #pragma unroll
        for (int nt = 0; nt < 8; ++nt) {
            const int col = nt * 16 + l16;
            const float bb = b1[col];
            const float w2 = W2[col];
            #pragma unroll
            for (int j = 0; j < 8; ++j)
                partial[j] += tanhf(acc[nt][j] + bb) * w2;
        }
        #pragma unroll
        for (int j = 0; j < 8; ++j) {
            float v = partial[j];
            v += __shfl_xor(v, 1, 32);
            v += __shfl_xor(v, 2, 32);
            v += __shfl_xor(v, 4, 32);
            v += __shfl_xor(v, 8, 32);   // reduced over the 16 lanes of this half
            if (l16 == 0) {
                int row = rowBase + j + 8 * half;   // flattened n*M + m
                atomicAdd(&score_sum[row % M_], v);
            }
        }
    }
}

// ------------------------------------------------- softmax over M scores
__global__ void k_softmax(const float* __restrict__ score_sum,
                          float* __restrict__ beta) {
    if (threadIdx.x == 0) {
        float w0 = score_sum[0] / (float)N_;
        float w1 = score_sum[1] / (float)N_;
        float w2 = score_sum[2] / (float)N_;
        float mx = fmaxf(w0, fmaxf(w1, w2));
        float e0 = expf(w0 - mx), e1 = expf(w1 - mx), e2 = expf(w2 - mx);
        float s = e0 + e1 + e2;
        beta[0] = e0 / s; beta[1] = e1 / s; beta[2] = e2 / s;
    }
}

// ----------------------------------------------- out = sum_m beta[m]*z[:,m,:]
__global__ void k_combine(const float* __restrict__ z,
                          const float* __restrict__ beta,
                          float* __restrict__ out) {
    int tid = blockIdx.x * blockDim.x + threadIdx.x;
    if (tid >= N_ * (D_ / 4)) return;
    int n = tid >> 5;                    // D/4 == 32
    int q = tid & 31;
    const float4* zr = (const float4*)(z + (size_t)n * M_ * D_);
    float b0 = beta[0], b1v = beta[1], b2 = beta[2];
    float4 a = zr[q], b = zr[32 + q], c = zr[64 + q];
    float4 o;
    o.x = b0 * a.x + b1v * b.x + b2 * c.x;
    o.y = b0 * a.y + b1v * b.y + b2 * c.y;
    o.z = b0 * a.z + b1v * b.z + b2 * c.z;
    o.w = b0 * a.w + b1v * b.w + b2 * c.w;
    ((float4*)out)[tid] = o;
}

extern "C" void kernel_launch(void* const* d_in, const int* in_sizes, int n_in,
                              void* d_out, int out_size, void* d_ws, size_t ws_size,
                              hipStream_t stream) {
    (void)in_sizes; (void)n_in; (void)out_size; (void)ws_size;
    const float* h     = (const float*)d_in[0];
    const int*   edges = (const int*)  d_in[1];
    const float* W1    = (const float*)d_in[2];
    const float* b1    = (const float*)d_in[3];
    const float* W2    = (const float*)d_in[4];
    float* out = (float*)d_out;

    // Workspace layout (floats, then bf16 tail)
    float* ws        = (float*)d_ws;
    float* deg_src   = ws;                                   // M*N
    float* deg_dst   = deg_src + (size_t)M_ * N_;            // M*N
    float* z         = deg_dst + (size_t)M_ * N_;            // N*M*D (16B aligned)
    float* score_sum = z + (size_t)N_ * M_ * D_;             // 4
    float* beta      = score_sum + 4;                        // 4
    __bf16* zbf      = (__bf16*)(beta + 4);                  // N*M*D bf16 (32B aligned)
    const size_t zero_count = 2 * (size_t)M_ * N_ + (size_t)N_ * M_ * D_ + 8;

    // 1) zero degree counters, z accumulator, score accumulators
    k_zero<<<4096, 256, 0, stream>>>(ws, zero_count);

    // 2) degree histograms (f32 atomics, exact for counts < 2^24)
    {
        long long t = (long long)M_ * E_;
        k_deg<<<(unsigned)((t + 255) / 256), 256, 0, stream>>>(edges, deg_src, deg_dst);
    }

    // 3) rsqrt(max(deg,1)) in place over both arrays (contiguous)
    {
        int t = 2 * M_ * N_;
        k_norm<<<(t + 255) / 256, 256, 0, stream>>>(deg_src, t);
    }

    // 4) gather/scatter-add: one wave32 per (metapath, edge)
    {
        long long waves = (long long)M_ * E_;                // 4.8M waves
        unsigned blocks = (unsigned)((waves + 7) / 8);       // 8 waves / 256-thr block
        k_scatter<<<blocks, 256, 0, stream>>>(h, edges, deg_src, z);
    }

    // 5) scale rows by dst norm + produce bf16 z for the WMMA GEMM
    {
        size_t t = (size_t)N_ * M_ * (D_ / 4);
        k_scale_dst<<<(unsigned)((t + 255) / 256), 256, 0, stream>>>(z, deg_dst, zbf);
    }

    // 6) WMMA attention-score GEMM (+tanh, dot W2, per-metapath reduce)
    //    persistent blocks: amortize the 32KB W1^T LDS staging over ~18 tiles
    k_attn_score<<<512, 256, 0, stream>>>(zbf, W1, b1, W2, score_sum);

    // 7) softmax over the M=3 scores
    k_softmax<<<1, 32, 0, stream>>>(score_sum, beta);

    // 8) weighted combine -> output [N, D]
    {
        int t = N_ * (D_ / 4);
        k_combine<<<(t + 255) / 256, 256, 0, stream>>>(z, beta, out);
    }
}
